// Graph_resnet_9405978378358
// MI455X (gfx1250) — compile-verified, matrix operand order checked by reading, not run on hardware
//
#include <hip/hip_runtime.h>
#include <hip/hip_bf16.h>

#define N_NODES  100000
#define N_HEDGES 20000
#define N_INC    1000000

typedef __attribute__((ext_vector_type(16))) __bf16 v16bf;
typedef __attribute__((ext_vector_type(4)))  __bf16 bf16x4;
typedef __attribute__((ext_vector_type(8)))  float  v8f;

// ---------------------------------------------------------------------------
// Generic GEMM  C[N,F] = concat(A0[N,K0], A1[N,K1]) @ W[K0+K1, F]
// bf16 inputs staged via LDS (float4 global loads -> bf16x4 LDS stores),
// f32 accumulate with v_wmma_f32_16x16x32_bf16.
// Block covers 64 rows x F cols; one 16x16 tile per wave (4*(F/16) waves).
// Compile-time <K0,K1,F> specialization removes all boundary branching.
// ---------------------------------------------------------------------------
#define BM  64
#define KC  64
#define AST (KC + 8)
#define WST (KC + 8)

template <int K0, int K1, int F>
__global__ __launch_bounds__(32 * 4 * (F / 16)) void
gemm_wmma(const float* __restrict__ A0, const float* __restrict__ A1,
          const float* __restrict__ W, float* __restrict__ C, int N) {
  static_assert(K0 % 4 == 0 && (K0 + K1) % 4 == 0, "4-elem groups never straddle");
  static_assert(F % 16 == 0 && F <= 80, "F tiles");
  constexpr int K = K0 + K1;
  __shared__ __bf16 Alds[BM * AST];   // [row][k]
  __shared__ __bf16 Wlds[F * WST];    // [col][k] (transposed)
  const int tid = threadIdx.x;
  constexpr int nThreads = 32 * 4 * (F / 16);
  const int wave = tid >> 5;
  const int lane = tid & 31;
  constexpr int nTilesN = F >> 4;
  const int mSub = wave / nTilesN;          // 0..3
  const int nSub = wave % nTilesN;          // 0..F/16-1
  const int rowBase = blockIdx.x * BM;

  v8f acc = {};

  for (int kc = 0; kc < K; kc += KC) {
    __syncthreads();

    // ---- Stage A chunk: BM x KC, one float4 -> bf16x4 per thread-iter -----
    for (int idx = tid; idx < BM * (KC / 4); idx += nThreads) {
      const int r  = idx >> 4;            // KC/4 == 16
      const int c4 = (idx & 15) << 2;
      const int grow = rowBase + r;
      const int gk = kc + c4;
      bf16x4 v = {};
      if (grow < N && gk < K) {
        const float* src = (K1 == 0 || gk < K0)
                               ? &A0[(size_t)grow * K0 + gk]
                               : &A1[(size_t)grow * K1 + (gk - K0)];
        float4 f = *(const float4*)src;
        v.x = (__bf16)f.x; v.y = (__bf16)f.y;
        v.z = (__bf16)f.z; v.w = (__bf16)f.w;
      }
      *(bf16x4*)&Alds[r * AST + c4] = v;  // 8B-aligned ds_store_b64
    }

    // ---- Stage W chunk transposed: float4 along n, scalar LDS stores ------
    for (int idx = tid; idx < (F >> 2) * KC; idx += nThreads) {
      const int c  = idx & (KC - 1);
      const int n4 = (idx >> 6) << 2;     // KC == 64
      const int gk = kc + c;
      float4 f = {0.0f, 0.0f, 0.0f, 0.0f};
      if (gk < K) f = *(const float4*)&W[(size_t)gk * F + n4];
      Wlds[(n4 + 0) * WST + c] = (__bf16)f.x;
      Wlds[(n4 + 1) * WST + c] = (__bf16)f.y;
      Wlds[(n4 + 2) * WST + c] = (__bf16)f.z;
      Wlds[(n4 + 3) * WST + c] = (__bf16)f.w;
    }
    __syncthreads();

#pragma unroll
    for (int ks = 0; ks < KC; ks += 32) {
      // A fragment: lane holds row M = lane%16, K = ks + (lane/16)*8 + {0..7, 16..23}
      const int m  = mSub * 16 + (lane & 15);
      const int kb = ks + ((lane >> 4) << 3);
      const __bf16* ap = &Alds[m * AST + kb];
      // B fragment: lane holds col N = lane%16, same K pattern
      const __bf16* bp = &Wlds[(nSub * 16 + (lane & 15)) * WST + kb];
      v16bf afrag, bfrag;
#pragma unroll
      for (int c2 = 0; c2 < 8; ++c2) {
        afrag[c2]     = ap[c2];
        afrag[c2 + 8] = ap[c2 + 16];
        bfrag[c2]     = bp[c2];
        bfrag[c2 + 8] = bp[c2 + 16];
      }
      acc = __builtin_amdgcn_wmma_f32_16x16x32_bf16(
          /*neg_a=*/false, afrag, /*neg_b=*/false, bfrag,
          /*c_mod=*/(short)0, acc, /*reuse_a=*/false, /*reuse_b=*/false);
    }
  }

  // C/D layout: lane = n + 16*(m/8), vgpr r -> m = r + 8*(lane/16)
  const int n    = nSub * 16 + (lane & 15);
  const int mOff = (lane >> 4) << 3;
#pragma unroll
  for (int r = 0; r < 8; ++r) {
    int row = rowBase + mSub * 16 + mOff + r;
    if (row < N) C[(size_t)row * F + n] = acc[r];
  }
}

// ---------------------------------------------------------------------------
// Degree counting + inversion
// ---------------------------------------------------------------------------
__global__ void deg_kernel(const int* __restrict__ ni, const int* __restrict__ he,
                           float* __restrict__ D, float* __restrict__ B, int n) {
  int i = blockIdx.x * blockDim.x + threadIdx.x;
  if (i >= n) return;
  atomicAdd(&D[ni[i]], 1.0f);
  atomicAdd(&B[he[i]], 1.0f);
}

__global__ void inv_kernel(float* __restrict__ p, int n) {
  int i = blockIdx.x * blockDim.x + threadIdx.x;
  if (i >= n) return;
  float v = p[i];
  p[i] = (v > 0.0f) ? 1.0f / v : 0.0f;
}

// ---------------------------------------------------------------------------
// Incidence scatters (float4-vectorized atomics; targets are L2-resident)
// ---------------------------------------------------------------------------
__global__ void scatter_edges(const float* __restrict__ g, const int* __restrict__ ni,
                              const int* __restrict__ he, float* __restrict__ e,
                              int nInc, int F) {
  const int C = F >> 2;
  long long idx = (long long)blockIdx.x * blockDim.x + threadIdx.x;
  if (idx >= (long long)nInc * C) return;
  int p = (int)(idx / C);
  int c = (int)(idx % C) << 2;
  int n = ni[p], h = he[p];
  float4 v = *(const float4*)&g[(size_t)n * F + c];
  float* dst = &e[(size_t)h * F + c];
  atomicAdd(dst + 0, v.x);
  atomicAdd(dst + 1, v.y);
  atomicAdd(dst + 2, v.z);
  atomicAdd(dst + 3, v.w);
}

__global__ void scatter_nodes(const float* __restrict__ e, const float* __restrict__ Binv,
                              const int* __restrict__ ni, const int* __restrict__ he,
                              float* __restrict__ o, int nInc, int F) {
  const int C = F >> 2;
  long long idx = (long long)blockIdx.x * blockDim.x + threadIdx.x;
  if (idx >= (long long)nInc * C) return;
  int p = (int)(idx / C);
  int c = (int)(idx % C) << 2;
  int n = ni[p], h = he[p];
  float s = Binv[h];
  float4 v = *(const float4*)&e[(size_t)h * F + c];
  float* dst = &o[(size_t)n * F + c];
  atomicAdd(dst + 0, v.x * s);
  atomicAdd(dst + 1, v.y * s);
  atomicAdd(dst + 2, v.z * s);
  atomicAdd(dst + 3, v.w * s);
}

// ---------------------------------------------------------------------------
// BatchNorm: pass 1 applies Dinv+bias in place and accumulates sum/sumsq
// (thread = channel -> coalesced across a row); pass 2 finalizes scale/shift;
// pass 3 fuses relu(bn(h)) + skip*Dinv + bs.
// ---------------------------------------------------------------------------
__global__ void bn_stats(float* __restrict__ h, const float* __restrict__ Dinv,
                         const float* __restrict__ bias, float* __restrict__ stats,
                         int N, int F, int rowsPerBlock) {
  int c = threadIdx.x;
  if (c >= F) return;
  int r0 = blockIdx.x * rowsPerBlock;
  int r1 = r0 + rowsPerBlock; if (r1 > N) r1 = N;
  float s = 0.0f, s2 = 0.0f, b = bias[c];
  for (int r = r0; r < r1; ++r) {
    float v = h[(size_t)r * F + c] * Dinv[r] + b;
    h[(size_t)r * F + c] = v;
    s += v; s2 += v * v;
  }
  atomicAdd(&stats[c], s);
  atomicAdd(&stats[128 + c], s2);
}

__global__ void bn_finalize(float* __restrict__ stats, const float* __restrict__ gamma,
                            const float* __restrict__ beta, int F, float invN) {
  int c = threadIdx.x;
  if (c >= F) return;
  float mu  = stats[c] * invN;
  float var = stats[128 + c] * invN - mu * mu;
  float inv = rsqrtf(var + 1e-5f);
  float sc  = gamma[c] * inv;
  stats[256 + c] = sc;
  stats[384 + c] = beta[c] - sc * mu;
}

__global__ void bn_apply_add(const float* __restrict__ h, const float* __restrict__ o2,
                             const float* __restrict__ Dinv, const float* __restrict__ bs,
                             const float* __restrict__ stats, float* __restrict__ xout,
                             int N, int F) {
  long long idx = (long long)blockIdx.x * blockDim.x + threadIdx.x;
  if (idx >= (long long)N * F) return;
  int r = (int)(idx / F), c = (int)(idx % F);
  float hb = fmaxf(h[idx] * stats[256 + c] + stats[384 + c], 0.0f);
  xout[idx] = hb + o2[idx] * Dinv[r] + bs[c];
}

__global__ void final_scale_bias(float* __restrict__ out, const float* __restrict__ Dinv,
                                 const float* __restrict__ bmix, int N, int F) {
  long long idx = (long long)blockIdx.x * blockDim.x + threadIdx.x;
  if (idx >= (long long)N * F) return;
  int r = (int)(idx / F), c = (int)(idx % F);
  out[idx] = out[idx] * Dinv[r] + bmix[c];
}

// ---------------------------------------------------------------------------
// Host-side hconv: GEMM -> zero e -> edge scatter -> zero dst -> node scatter
// ---------------------------------------------------------------------------
template <int K0, int K1, int F>
static void launch_hconv(const float* A0, const float* A1, const float* W,
                         float* dst, float* g, float* ebuf, const float* Binv,
                         const int* ni, const int* he, hipStream_t stream) {
  dim3 blk(32 * 4 * (F / 16));
  dim3 grd((N_NODES + BM - 1) / BM);
  gemm_wmma<K0, K1, F><<<grd, blk, 0, stream>>>(A0, A1, W, g, N_NODES);
  hipMemsetAsync(ebuf, 0, (size_t)N_HEDGES * F * sizeof(float), stream);
  long long tot = (long long)N_INC * (F / 4);
  scatter_edges<<<(unsigned)((tot + 255) / 256), 256, 0, stream>>>(g, ni, he, ebuf, N_INC, F);
  hipMemsetAsync(dst, 0, (size_t)N_NODES * F * sizeof(float), stream);
  scatter_nodes<<<(unsigned)((tot + 255) / 256), 256, 0, stream>>>(ebuf, Binv, ni, he, dst, N_INC, F);
}

// ---------------------------------------------------------------------------
extern "C" void kernel_launch(void* const* d_in, const int* in_sizes, int n_in,
                              void* d_out, int out_size, void* d_ws, size_t ws_size,
                              hipStream_t stream) {
  const float* x     = (const float*)d_in[0];
  const int*   eidx  = (const int*)d_in[1];
  const float* w1    = (const float*)d_in[2];
  const float* b1    = (const float*)d_in[3];
  const float* sw1   = (const float*)d_in[4];
  const float* sb1   = (const float*)d_in[5];
  const float* Wc    = (const float*)d_in[6];
  const float* bc    = (const float*)d_in[7];
  const float* Wsk   = (const float*)d_in[8];
  const float* bsk   = (const float*)d_in[9];
  const float* gamma = (const float*)d_in[10];
  const float* beta  = (const float*)d_in[11];
  const float* wmix  = (const float*)d_in[12];
  const float* bmix  = (const float*)d_in[13];
  float* out = (float*)d_out;

  const int* ni = eidx;
  const int* he = eidx + N_INC;

  // Workspace carve-out (~167 MB)
  float* ws   = (float*)d_ws;
  float* Dinv = ws;  ws += N_NODES;
  float* Binv = ws;  ws += N_HEDGES;
  float* st   = ws;  ws += 512;
  float* g    = ws;  ws += (size_t)N_NODES * 80;   // GEMM output (reused)
  float* ebuf = ws;  ws += (size_t)N_HEDGES * 80;  // hyperedge accumulator
  float* o1   = ws;  ws += (size_t)N_NODES * 80;   // conv path
  float* o2   = ws;  ws += (size_t)N_NODES * 80;   // skip path
  float* xA   = ws;  ws += (size_t)N_NODES * 80;
  float* xB   = ws;  ws += (size_t)N_NODES * 80;

  // Degrees
  hipMemsetAsync(Dinv, 0, (size_t)(N_NODES + N_HEDGES) * sizeof(float), stream);
  deg_kernel<<<(N_INC + 255) / 256, 256, 0, stream>>>(ni, he, Dinv, Binv, N_INC);
  inv_kernel<<<(N_NODES + 255) / 256, 256, 0, stream>>>(Dinv, N_NODES);
  inv_kernel<<<(N_HEDGES + 255) / 256, 256, 0, stream>>>(Binv, N_HEDGES);

  const float* xin = x;
  float* xcur = xA;
  float* xnext = xB;

  for (int l = 0; l < 7; ++l) {
    const float* W  = (l == 0) ? w1  : Wc  + (size_t)(l - 1) * 80 * 80;
    const float* b  = (l == 0) ? b1  : bc  + (size_t)(l - 1) * 80;
    const float* Wk = (l == 0) ? sw1 : Wsk + (size_t)(l - 1) * 80 * 80;
    const float* bk = (l == 0) ? sb1 : bsk + (size_t)(l - 1) * 80;

    if (l == 0) {
      launch_hconv<96, 0, 80>(xin, nullptr, W,  o1, g, ebuf, Binv, ni, he, stream);
      launch_hconv<96, 0, 80>(xin, nullptr, Wk, o2, g, ebuf, Binv, ni, he, stream);
    } else {
      launch_hconv<80, 0, 80>(xin, nullptr, W,  o1, g, ebuf, Binv, ni, he, stream);
      launch_hconv<80, 0, 80>(xin, nullptr, Wk, o2, g, ebuf, Binv, ni, he, stream);
    }

    hipMemsetAsync(st, 0, 512 * sizeof(float), stream);
    bn_stats<<<(N_NODES + 127) / 128, 128, 0, stream>>>(o1, Dinv, b, st, N_NODES, 80, 128);
    bn_finalize<<<1, 128, 0, stream>>>(st, gamma + (size_t)l * 80, beta + (size_t)l * 80,
                                       80, 1.0f / (float)N_NODES);
    long long tot = (long long)N_NODES * 80;
    bn_apply_add<<<(unsigned)((tot + 255) / 256), 256, 0, stream>>>(
        o1, o2, Dinv, bk, st, xnext, N_NODES, 80);

    float* t = xcur; xcur = xnext; xnext = t;
    xin = xcur;
  }

  // Final mix over virtual concat [xcur | x0] (K = 176) -> out[N,48]
  launch_hconv<80, 96, 48>(xcur, x, wmix, out, g, ebuf, Binv, ni, he, stream);
  long long tot = (long long)N_NODES * 48;
  final_scale_bias<<<(unsigned)((tot + 255) / 256), 256, 0, stream>>>(out, Dinv, bmix, N_NODES, 48);
}
